// ProteinFeatureExtractor_67688684585486
// MI455X (gfx1250) — matrix operand order repaired and checked by reading, not compiled, not run on hardware
//
#include <hip/hip_runtime.h>
#include <hip/hip_bf16.h>
#include <math.h>

#define N_NODES 30000
#define N_EDGES 480000
#define N_GRAPHS 32
#define HDIM 512
#define FUSED_DIM 768
#define KPROJ_PAD 96
#define BN_RSQRT 0.99999500003749974f  // 1/sqrt(1+1e-5)

typedef __attribute__((ext_vector_type(16))) __bf16 v16bf;
typedef __attribute__((ext_vector_type(8)))  float  v8f;
typedef __attribute__((ext_vector_type(8)))  unsigned short v8u16;

// epilogue flags
#define F_BN1  1
#define F_RES  2
#define F_BN2  4
#define F_RELU 8
#define F_TANH 16

// ---------------------------------------------------------------------------
// Weight convert + transpose: W[K][N] f32 -> WT[N][Kpad] bf16 (zero-padded K)
// ---------------------------------------------------------------------------
__global__ void wconvT_kernel(const float* __restrict__ W, __bf16* __restrict__ WT,
                              int K, int N, int Kpad) {
    int idx = blockIdx.x * 256 + threadIdx.x;
    int total = N * Kpad;
    if (idx >= total) return;
    int o = idx / Kpad, k = idx - o * Kpad;
    float v = (k < K) ? W[(size_t)k * N + o] : 0.0f;
    WT[(size_t)o * Kpad + k] = (__bf16)v;
}

// ---------------------------------------------------------------------------
// Main WMMA GEMM: C[M x N] = epi( A[M x K](f32) @ WT[N x K]^T(bf16) )
// one wave = one 16x64 output strip (4 accumulators share one A fragment);
// K multiple of 32; M multiple of 16; N multiple of 64
// ---------------------------------------------------------------------------
__global__ __launch_bounds__(256)
void gemm_wmma_kernel(const float* __restrict__ A, const __bf16* __restrict__ WT,
                      const float* __restrict__ bias,
                      const float* __restrict__ bn1g, const float* __restrict__ bn1b,
                      const float* __restrict__ resid,
                      const float* __restrict__ bn2g, const float* __restrict__ bn2b,
                      float* __restrict__ C,
                      int M, int N, int K, int ldc, int cofs, int flags) {
    const int lane = threadIdx.x & 31;
    const int wave = blockIdx.x * (blockDim.x >> 5) + (threadIdx.x >> 5);
    const int tilesN = N >> 6;                       // strips of 64 columns
    const int tilesTotal = (M >> 4) * tilesN;
    if (wave >= tilesTotal) return;                  // wave-uniform: EXEC stays all-ones
    const int tm = wave / tilesN;
    const int tn = wave - tm * tilesN;
    const int half = lane >> 4;                      // 0: lanes 0-15, 1: lanes 16-31
    const int l16  = lane & 15;

    const float*  arow  = A  + (size_t)(tm * 16 + l16) * K;
    const __bf16* bbase = WT + (size_t)(tn * 64 + l16) * K + half * 16;
    const size_t  bstr  = (size_t)16 * K;            // 16 columns per sub-tile

    v8f acc0 = {}, acc1 = {}, acc2 = {}, acc3 = {};
    for (int k0 = 0; k0 < K; k0 += 32) {
        if (k0 + 32 < K) {
            __builtin_prefetch((const void*)(arow + k0 + 32), 0, 3);
            __builtin_prefetch((const void*)(bbase + k0 + 32), 0, 3);
        }
        // A fragment: f32 rows, convert to bf16 in-register (shared by 4 WMMAs)
        const float4 fa = *(const float4*)(arow + k0 + half * 16 + 0);
        const float4 fb = *(const float4*)(arow + k0 + half * 16 + 4);
        const float4 fc = *(const float4*)(arow + k0 + half * 16 + 8);
        const float4 fd = *(const float4*)(arow + k0 + half * 16 + 12);
        float tmp[16] = {fa.x, fa.y, fa.z, fa.w, fb.x, fb.y, fb.z, fb.w,
                         fc.x, fc.y, fc.z, fc.w, fd.x, fd.y, fd.z, fd.w};
        v16bf a;
#pragma unroll
        for (int j = 0; j < 16; ++j) a[j] = (__bf16)tmp[j];
        // 4 B fragments: contiguous 16 bf16 each from transposed weights
        union { v16bf bf; v8u16 us[2]; } b0, b1, b2, b3;
        b0.us[0] = *(const v8u16*)(bbase + 0 * bstr + k0);
        b0.us[1] = *(const v8u16*)(bbase + 0 * bstr + k0 + 8);
        b1.us[0] = *(const v8u16*)(bbase + 1 * bstr + k0);
        b1.us[1] = *(const v8u16*)(bbase + 1 * bstr + k0 + 8);
        b2.us[0] = *(const v8u16*)(bbase + 2 * bstr + k0);
        b2.us[1] = *(const v8u16*)(bbase + 2 * bstr + k0 + 8);
        b3.us[0] = *(const v8u16*)(bbase + 3 * bstr + k0);
        b3.us[1] = *(const v8u16*)(bbase + 3 * bstr + k0 + 8);
        acc0 = __builtin_amdgcn_wmma_f32_16x16x32_bf16(false, a, false, b0.bf, (short)0, acc0, false, false);
        acc1 = __builtin_amdgcn_wmma_f32_16x16x32_bf16(false, a, false, b1.bf, (short)0, acc1, false, false);
        acc2 = __builtin_amdgcn_wmma_f32_16x16x32_bf16(false, a, false, b2.bf, (short)0, acc2, false, false);
        acc3 = __builtin_amdgcn_wmma_f32_16x16x32_bf16(false, a, false, b3.bf, (short)0, acc3, false, false);
    }

    v8f accs[4] = {acc0, acc1, acc2, acc3};
#pragma unroll
    for (int j = 0; j < 4; ++j) {
        const int col = tn * 64 + j * 16 + l16;
        const float bia = bias ? bias[col] : 0.0f;
        float g1 = 0.f, s1 = 0.f, g2 = 0.f, s2 = 0.f;
        if (flags & F_BN1) { g1 = bn1g[col] * BN_RSQRT; s1 = bn1b[col]; }
        if (flags & F_BN2) { g2 = bn2g[col] * BN_RSQRT; s2 = bn2b[col]; }
#pragma unroll
        for (int i = 0; i < 8; ++i) {
            const int row = tm * 16 + half * 8 + i;
            float v = accs[j][i] + bia;
            if (flags & F_BN1) v = v * g1 + s1;
            if (flags & F_RES) v += resid[(size_t)row * N + col];
            if (flags & F_BN2) v = v * g2 + s2;
            if (flags & F_RELU) v = fmaxf(v, 0.0f);
            if (flags & F_TANH) v = tanhf(v);
            C[(size_t)row * ldc + cofs + col] = v;
        }
    }
}

// ---------------------------------------------------------------------------
// Small elementwise / irregular kernels
// ---------------------------------------------------------------------------
__global__ void posenc1_kernel(const float* __restrict__ pos,
                               const float* __restrict__ w, const float* __restrict__ b,
                               float* __restrict__ out) {
    int idx = blockIdx.x * 256 + threadIdx.x;
    if (idx >= N_NODES * 64) return;
    int n = idx >> 6, c = idx & 63;
    float acc = b[c];
#pragma unroll
    for (int k = 0; k < 3; ++k) acc += pos[n * 3 + k] * w[k * 64 + c];
    out[idx] = fmaxf(acc, 0.0f);
}

__global__ void posenc2_kernel(const float* __restrict__ in,
                               const float* __restrict__ w, const float* __restrict__ b,
                               float* __restrict__ hin) {
    int idx = blockIdx.x * 256 + threadIdx.x;
    if (idx >= N_NODES * 64) return;
    int n = idx >> 6, c = idx & 63;
    float acc = b[c];
    for (int k = 0; k < 64; ++k) acc += in[n * 64 + k] * w[k * 64 + c];
    hin[(size_t)n * KPROJ_PAD + 6 + c] = fmaxf(acc, 0.0f);
}

__global__ void fill_hin_kernel(const float* __restrict__ x, float* __restrict__ hin) {
    int idx = blockIdx.x * 256 + threadIdx.x;
    if (idx >= N_NODES * KPROJ_PAD) return;
    int n = idx / KPROJ_PAD, c = idx - n * KPROJ_PAD;
    hin[idx] = (c < 6) ? x[n * 6 + c] : 0.0f;  // cols 6..69 overwritten by posenc2
}

__global__ void copyf_kernel(const float* __restrict__ src, float* __restrict__ dst, int n) {
    int idx = blockIdx.x * 256 + threadIdx.x;
    if (idx < n) dst[idx] = src[idx];
}

__global__ void zerof_kernel(float* __restrict__ p, int n) {
    int idx = blockIdx.x * 256 + threadIdx.x;
    if (idx < n) p[idx] = 0.0f;
}

__global__ void edge_scatter_kernel(const float* __restrict__ h,
                                    const int* __restrict__ src, const int* __restrict__ dst,
                                    float* __restrict__ agg) {
    const int CH = HDIM / 4;  // 128 float4 chunks per row
    long long t = (long long)blockIdx.x * 256 + threadIdx.x;
    long long total = (long long)N_EDGES * CH;
    if (t >= total) return;
    int e = (int)(t / CH);
    int c = (int)(t - (long long)e * CH) * 4;
    int s = src[e], d = dst[e];
    const float4 v = *(const float4*)(h + (size_t)s * HDIM + c);
    float* o = agg + (size_t)d * HDIM + c;
    atomicAdd(o + 0, v.x); atomicAdd(o + 1, v.y);
    atomicAdd(o + 2, v.z); atomicAdd(o + 3, v.w);
}

__global__ void attn_score_kernel(const float* __restrict__ ah,
                                  const float* __restrict__ w2, const float* __restrict__ b2,
                                  float* __restrict__ a) {
    int n = blockIdx.x * 256 + threadIdx.x;
    if (n >= N_NODES) return;
    float acc = b2[0];
    const float* r = ah + (size_t)n * HDIM;
    for (int k = 0; k < HDIM; ++k) acc += r[k] * w2[k];
    a[n] = acc;
}

__global__ void reduce_max_kernel(const float* __restrict__ a, int n, float* __restrict__ out) {
    __shared__ float sd[256];
    float m = -3.402823e38f;
    for (int i = blockIdx.x * 256 + threadIdx.x; i < n; i += gridDim.x * 256)
        m = fmaxf(m, a[i]);
    sd[threadIdx.x] = m; __syncthreads();
    for (int s = 128; s > 0; s >>= 1) {
        if (threadIdx.x < s) sd[threadIdx.x] = fmaxf(sd[threadIdx.x], sd[threadIdx.x + s]);
        __syncthreads();
    }
    if (threadIdx.x == 0) out[blockIdx.x] = sd[0];
}

__global__ void reduce_sum_kernel(const float* __restrict__ a, int n, float* __restrict__ out) {
    __shared__ float sd[256];
    float m = 0.0f;
    for (int i = blockIdx.x * 256 + threadIdx.x; i < n; i += gridDim.x * 256) m += a[i];
    sd[threadIdx.x] = m; __syncthreads();
    for (int s = 128; s > 0; s >>= 1) {
        if (threadIdx.x < s) sd[threadIdx.x] += sd[threadIdx.x + s];
        __syncthreads();
    }
    if (threadIdx.x == 0) out[blockIdx.x] = sd[0];
}

__global__ void exp_partial_kernel(const float* __restrict__ a, const float* __restrict__ mx,
                                   float* __restrict__ w, int n, float* __restrict__ part) {
    __shared__ float sd[256];
    const float m = mx[0];
    float acc = 0.0f;
    for (int i = blockIdx.x * 256 + threadIdx.x; i < n; i += gridDim.x * 256) {
        float e = __expf(a[i] - m);
        w[i] = e;
        acc += e;
    }
    sd[threadIdx.x] = acc; __syncthreads();
    for (int s = 128; s > 0; s >>= 1) {
        if (threadIdx.x < s) sd[threadIdx.x] += sd[threadIdx.x + s];
        __syncthreads();
    }
    if (threadIdx.x == 0) part[blockIdx.x] = sd[0];
}

__global__ void counts_kernel(const int* __restrict__ batch, float* __restrict__ counts) {
    int n = blockIdx.x * 256 + threadIdx.x;
    if (n < N_NODES) atomicAdd(&counts[batch[n]], 1.0f);
}

__global__ void pool_scatter_kernel(const float* __restrict__ fused, const float* __restrict__ w,
                                    const float* __restrict__ ssum, const int* __restrict__ batch,
                                    float* __restrict__ wsum, float* __restrict__ gmean,
                                    float* __restrict__ gmax) {
    const int CH = FUSED_DIM / 4;  // 192
    long long t = (long long)blockIdx.x * 256 + threadIdx.x;
    long long total = (long long)N_NODES * CH;
    if (t >= total) return;
    int n = (int)(t / CH);
    int c = (int)(t - (long long)n * CH) * 4;
    int g = batch[n];
    float wn = w[n] / ssum[0];
    const float4 v = *(const float4*)(fused + (size_t)n * FUSED_DIM + c);
    float* ws = wsum  + (size_t)g * FUSED_DIM + c;
    float* gm = gmean + (size_t)g * FUSED_DIM + c;
    float* gx = gmax  + (size_t)g * FUSED_DIM + c;
    atomicAdd(ws + 0, v.x * wn); atomicAdd(ws + 1, v.y * wn);
    atomicAdd(ws + 2, v.z * wn); atomicAdd(ws + 3, v.w * wn);
    atomicAdd(gm + 0, v.x); atomicAdd(gm + 1, v.y);
    atomicAdd(gm + 2, v.z); atomicAdd(gm + 3, v.w);
    // fused >= 0 (post-ReLU): int-pattern max is order-preserving vs 0-init
    atomicMax((int*)(gx + 0), __float_as_int(v.x));
    atomicMax((int*)(gx + 1), __float_as_int(v.y));
    atomicMax((int*)(gx + 2), __float_as_int(v.z));
    atomicMax((int*)(gx + 3), __float_as_int(v.w));
}

__global__ void assemble_pooled_kernel(const float* __restrict__ wsum, const float* __restrict__ gmean,
                                       const float* __restrict__ gmax, const float* __restrict__ counts,
                                       float* __restrict__ pooled) {
    int idx = blockIdx.x * 256 + threadIdx.x;
    if (idx >= N_GRAPHS * 3 * FUSED_DIM) return;
    int g = idx / (3 * FUSED_DIM), c = idx - g * 3 * FUSED_DIM;
    float cnt = counts[g];
    float denom = fmaxf(cnt, 1.0f);
    float v;
    if (c < FUSED_DIM)            v = wsum[g * FUSED_DIM + c] / denom;
    else if (c < 2 * FUSED_DIM)   v = gmean[g * FUSED_DIM + (c - FUSED_DIM)] / denom;
    else                          v = (cnt > 0.0f) ? gmax[g * FUSED_DIM + (c - 2 * FUSED_DIM)] : 0.0f;
    pooled[idx] = v;
}

// ---------------------------------------------------------------------------
static inline int ceil_div(long long a, int b) { return (int)((a + b - 1) / b); }

static void launch_gemm(hipStream_t stream, const float* A, const __bf16* WT,
                        const float* bias, const float* bn1g, const float* bn1b,
                        const float* resid, const float* bn2g, const float* bn2b,
                        float* C, int M, int N, int K, int ldc, int cofs, int flags) {
    int waves = (M / 16) * (N / 64);   // one wave per 16x64 strip
    int blocks = (waves + 7) / 8;      // 8 waves per 256-thread block
    gemm_wmma_kernel<<<blocks, 256, 0, stream>>>(A, WT, bias, bn1g, bn1b, resid,
                                                 bn2g, bn2b, C, M, N, K, ldc, cofs, flags);
}

extern "C" void kernel_launch(void* const* d_in, const int* in_sizes, int n_in,
                              void* d_out, int out_size, void* d_ws, size_t ws_size,
                              hipStream_t stream) {
    const float* x         = (const float*)d_in[0];
    const int*   edge      = (const int*)d_in[1];
    const float* positions = (const float*)d_in[2];
    const int*   batch     = (const int*)d_in[3];
    const float* pos_w1 = (const float*)d_in[4];   const float* pos_b1 = (const float*)d_in[5];
    const float* pos_w2 = (const float*)d_in[6];   const float* pos_b2 = (const float*)d_in[7];
    const float* proj_w = (const float*)d_in[8];   const float* proj_b = (const float*)d_in[9];
    const float* proj_g = (const float*)d_in[10];  const float* proj_bb= (const float*)d_in[11];
    const float* res_w  = (const float*)d_in[12];  const float* res_b  = (const float*)d_in[13];
    const float* gin_w1 = (const float*)d_in[14];  const float* gin_b1 = (const float*)d_in[15];
    const float* gin_w2 = (const float*)d_in[16];  const float* gin_b2 = (const float*)d_in[17];
    const float* gin_g  = (const float*)d_in[18];  const float* gin_bb = (const float*)d_in[19];
    const float* out_g  = (const float*)d_in[20];  const float* out_bb = (const float*)d_in[21];
    const float* fus_w  = (const float*)d_in[22];  const float* fus_b  = (const float*)d_in[23];
    const float* attn_w1= (const float*)d_in[24];  const float* attn_b1= (const float*)d_in[25];
    const float* attn_w2= (const float*)d_in[26];  const float* attn_b2= (const float*)d_in[27];
    const float* out_w1 = (const float*)d_in[28];  const float* out_b1 = (const float*)d_in[29];
    const float* out_w2 = (const float*)d_in[30];  const float* out_b2 = (const float*)d_in[31];

    const int* esrc = edge;
    const int* edst = edge + N_EDGES;

    // ---- workspace bump allocator ----
    char* base = (char*)d_ws;
    size_t off = 0;
    auto alloc = [&](size_t bytes) -> char* {
        char* p = base + off;
        off = (off + bytes + 255) & ~(size_t)255;
        return p;
    };
    // bf16 transposed weights
    __bf16* wt_proj = (__bf16*)alloc((size_t)HDIM * KPROJ_PAD * 2);
    __bf16* wt_res  = (__bf16*)alloc((size_t)HDIM * HDIM * 2);
    __bf16* wt_g1   = (__bf16*)alloc((size_t)5 * HDIM * HDIM * 2);
    __bf16* wt_g2   = (__bf16*)alloc((size_t)5 * HDIM * HDIM * 2);
    __bf16* wt_fus  = (__bf16*)alloc((size_t)3 * 256 * HDIM * 2);
    __bf16* wt_at1  = (__bf16*)alloc((size_t)HDIM * FUSED_DIM * 2);
    __bf16* wt_o1   = (__bf16*)alloc((size_t)1024 * 2304 * 2);
    __bf16* wt_o2   = (__bf16*)alloc((size_t)1024 * 1024 * 2);
    // f32 activations
    float* pos64  = (float*)alloc((size_t)N_NODES * 64 * 4);
    float* hin    = (float*)alloc((size_t)N_NODES * KPROJ_PAD * 4);
    float* h      = (float*)alloc((size_t)N_NODES * HDIM * 4);
    float* hres   = (float*)alloc((size_t)N_NODES * HDIM * 4);
    float* agg    = (float*)alloc((size_t)N_NODES * HDIM * 4);   // reused as attnh later
    float* z1     = (float*)alloc((size_t)N_NODES * HDIM * 4);
    float* fused  = (float*)alloc((size_t)N_NODES * FUSED_DIM * 4);
    float* ascore = (float*)alloc((size_t)N_NODES * 4);
    float* wexp   = (float*)alloc((size_t)N_NODES * 4);
    float* part   = (float*)alloc(256 * 4);
    float* scal   = (float*)alloc(64 * 4);                        // [0]=max [1]=sum
    float* wsum   = (float*)alloc((size_t)N_GRAPHS * FUSED_DIM * 4);
    float* gmean  = (float*)alloc((size_t)N_GRAPHS * FUSED_DIM * 4);
    float* gmax   = (float*)alloc((size_t)N_GRAPHS * FUSED_DIM * 4);
    float* counts = (float*)alloc((size_t)N_GRAPHS * 4);
    float* pooled = (float*)alloc((size_t)N_GRAPHS * 3 * FUSED_DIM * 4);
    float* o1     = (float*)alloc((size_t)N_GRAPHS * 1024 * 4);
    float* outp   = (float*)d_out;

    // ---- weight conversion (f32 -> bf16, transposed, K-padded for proj) ----
    auto wconv = [&](const float* W, __bf16* WT, int K, int N, int Kpad) {
        wconvT_kernel<<<ceil_div((long long)N * Kpad, 256), 256, 0, stream>>>(W, WT, K, N, Kpad);
    };
    wconv(proj_w, wt_proj, 70, HDIM, KPROJ_PAD);
    wconv(res_w, wt_res, HDIM, HDIM, HDIM);
    for (int i = 0; i < 5; ++i) {
        wconv(gin_w1 + (size_t)i * HDIM * HDIM, wt_g1 + (size_t)i * HDIM * HDIM, HDIM, HDIM, HDIM);
        wconv(gin_w2 + (size_t)i * HDIM * HDIM, wt_g2 + (size_t)i * HDIM * HDIM, HDIM, HDIM, HDIM);
    }
    for (int j = 0; j < 3; ++j)
        wconv(fus_w + (size_t)j * HDIM * 256, wt_fus + (size_t)j * 256 * HDIM, HDIM, 256, HDIM);
    wconv(attn_w1, wt_at1, FUSED_DIM, HDIM, FUSED_DIM);
    wconv(out_w1, wt_o1, 2304, 1024, 2304);
    wconv(out_w2, wt_o2, 1024, 1024, 1024);

    // ---- positional encoder + input assembly ----
    posenc1_kernel<<<ceil_div((long long)N_NODES * 64, 256), 256, 0, stream>>>(positions, pos_w1, pos_b1, pos64);
    fill_hin_kernel<<<ceil_div((long long)N_NODES * KPROJ_PAD, 256), 256, 0, stream>>>(x, hin);
    posenc2_kernel<<<ceil_div((long long)N_NODES * 64, 256), 256, 0, stream>>>(pos64, pos_w2, pos_b2, hin);

    // ---- projection: h = relu(bn(hin @ proj_w + b)) ----
    launch_gemm(stream, hin, wt_proj, proj_b, proj_g, proj_bb, nullptr, nullptr, nullptr,
                h, N_NODES, HDIM, KPROJ_PAD, HDIM, 0, F_BN1 | F_RELU);
    // ---- residual branch ----
    launch_gemm(stream, h, wt_res, res_b, nullptr, nullptr, nullptr, nullptr, nullptr,
                hres, N_NODES, HDIM, HDIM, HDIM, 0, 0);

    // ---- GIN layers ----
    const int NH = N_NODES * HDIM;
    for (int i = 0; i < 5; ++i) {
        copyf_kernel<<<ceil_div(NH, 256), 256, 0, stream>>>(h, agg, NH);
        edge_scatter_kernel<<<ceil_div((long long)N_EDGES * (HDIM / 4), 256), 256, 0, stream>>>(h, esrc, edst, agg);
        launch_gemm(stream, agg, wt_g1 + (size_t)i * HDIM * HDIM,
                    gin_b1 + i * HDIM, nullptr, nullptr, nullptr, nullptr, nullptr,
                    z1, N_NODES, HDIM, HDIM, HDIM, 0, F_RELU);
        int fl = F_BN1 | F_BN2 | F_RELU | (i == 0 ? F_RES : 0);
        launch_gemm(stream, z1, wt_g2 + (size_t)i * HDIM * HDIM,
                    gin_b2 + i * HDIM, gin_g + i * HDIM, gin_bb + i * HDIM,
                    (i == 0 ? hres : nullptr), out_g + i * HDIM, out_bb + i * HDIM,
                    h, N_NODES, HDIM, HDIM, HDIM, 0, fl);
        if (i >= 2) {
            int j = i - 2;
            launch_gemm(stream, h, wt_fus + (size_t)j * 256 * HDIM,
                        fus_b + j * 256, nullptr, nullptr, nullptr, nullptr, nullptr,
                        fused, N_NODES, 256, HDIM, FUSED_DIM, j * 256, F_RELU);
        }
    }

    // ---- attention pooling (global softmax over nodes) ----
    float* attnh = agg;  // reuse
    launch_gemm(stream, fused, wt_at1, attn_b1, nullptr, nullptr, nullptr, nullptr, nullptr,
                attnh, N_NODES, HDIM, FUSED_DIM, HDIM, 0, F_TANH);
    attn_score_kernel<<<ceil_div(N_NODES, 256), 256, 0, stream>>>(attnh, attn_w2, attn_b2, ascore);
    reduce_max_kernel<<<128, 256, 0, stream>>>(ascore, N_NODES, part);
    reduce_max_kernel<<<1, 256, 0, stream>>>(part, 128, scal + 0);
    exp_partial_kernel<<<128, 256, 0, stream>>>(ascore, scal + 0, wexp, N_NODES, part);
    reduce_sum_kernel<<<1, 256, 0, stream>>>(part, 128, scal + 1);

    int zn = N_GRAPHS * FUSED_DIM;
    zerof_kernel<<<ceil_div(zn, 256), 256, 0, stream>>>(wsum, zn);
    zerof_kernel<<<ceil_div(zn, 256), 256, 0, stream>>>(gmean, zn);
    zerof_kernel<<<ceil_div(zn, 256), 256, 0, stream>>>(gmax, zn);
    zerof_kernel<<<1, 256, 0, stream>>>(counts, N_GRAPHS);
    counts_kernel<<<ceil_div(N_NODES, 256), 256, 0, stream>>>(batch, counts);
    pool_scatter_kernel<<<ceil_div((long long)N_NODES * (FUSED_DIM / 4), 256), 256, 0, stream>>>(
        fused, wexp, scal + 1, batch, wsum, gmean, gmax);
    assemble_pooled_kernel<<<ceil_div(N_GRAPHS * 3 * FUSED_DIM, 256), 256, 0, stream>>>(
        wsum, gmean, gmax, counts, pooled);

    // ---- output MLP (32 rows; WMMA: 2x16 strips and 2x16 strips) ----
    launch_gemm(stream, pooled, wt_o1, out_b1, nullptr, nullptr, nullptr, nullptr, nullptr,
                o1, N_GRAPHS, 1024, 2304, 1024, 0, F_RELU);
    launch_gemm(stream, o1, wt_o2, out_b2, nullptr, nullptr, nullptr, nullptr, nullptr,
                outp, N_GRAPHS, 1024, 1024, 1024, 0, F_RELU);
}
